// DeepseekV3Attention_39797166965132
// MI455X (gfx1250) — matrix operand order repaired and checked by reading, not compiled, not run on hardware
//
#include <hip/hip_runtime.h>

// ---------------- problem constants (match reference) ----------------
#define B_SZ      32
#define H_SZ      128
#define NOPE_D    128
#define ROPE_D    64
#define V_D       128
#define LORA      512
#define CACHE_LEN 4096
#define START_POS 4095
#define T_TOTAL   4096          // START_POS + QLEN
#define DKV       576           // LORA + ROPE_D
#define SCALE_F   0.07216878364870322f   // (128+64)^-0.5

#define HG        16            // heads per workgroup (WMMA M)
#define NHG       (H_SZ / HG)   // 8 head-groups
#define SPLIT_T   2             // T splits per (b,hg)
#define T_PER_SP  (T_TOTAL / SPLIT_T)   // 2048
#define TT        64            // timesteps per tile
#define TILES     (T_PER_SP / TT)       // 32

// padded LDS strides (dwords) to avoid bank conflicts: 292 % 64 = 36, gcd(36,64)=4
#define RQK 292                 // row stride of sK / sQ in dwords (584 halfwords)
#define RP  36                  // row stride of sP in dwords (72 halfwords)

// ---------------- workspace layout (bytes) ----------------
#define OFF_QCOMB 0                                     // bf16 [B][H][576]
#define SZ_QCOMB  ((size_t)B_SZ * H_SZ * DKV * 2)       // 4,718,592
#define OFF_KPEN  (OFF_QCOMB + SZ_QCOMB)                // f32 [B][64]
#define SZ_KPEN   ((size_t)B_SZ * ROPE_D * 4)
#define OFF_M     (OFF_KPEN + SZ_KPEN)                  // f32 [B*NHG*SPLIT][16]
#define SZ_ML     ((size_t)B_SZ * NHG * SPLIT_T * HG * 4)
#define OFF_L     (OFF_M + SZ_ML)
#define OFF_ACC   (OFF_L + SZ_ML)                       // f32 [B*NHG*SPLIT][16][512]
#define SZ_ACC    ((size_t)B_SZ * NHG * SPLIT_T * HG * LORA * 4)

// ---------------- WMMA types ----------------
typedef __attribute__((ext_vector_type(16))) __bf16 v16bf;
typedef __attribute__((ext_vector_type(8)))  float  v8f;
typedef __attribute__((ext_vector_type(4)))  unsigned int u32x4;
union __attribute__((aligned(16))) Frag { unsigned int u[8]; u32x4 q[2]; v16bf v; };

__device__ __forceinline__ unsigned short f2bf(float f) {
    union { float f; unsigned u; } x; x.f = f;
    unsigned r = x.u + 0x7FFFu + ((x.u >> 16) & 1u);   // RNE
    return (unsigned short)(r >> 16);
}

// =====================================================================
// Kernel 1: q_lat = q_nope · w_ukv (per head), RoPE(q_pe, k_pe),
//           emit bf16 Q-combined [B,H,576] and f32 roped k_pe [B,64].
// grid = B*H blocks, 128 threads.
// =====================================================================
__global__ void prep_q_kernel(const float* __restrict__ q_nope,
                              const float* __restrict__ q_pe,
                              const float* __restrict__ k_pe,
                              const int*   __restrict__ position_ids,
                              const float* __restrict__ sin_t,
                              const float* __restrict__ cos_t,
                              const float* __restrict__ wkv_b,
                              unsigned short* __restrict__ qcomb,
                              float* __restrict__ kpe_new) {
    const int b = blockIdx.x / H_SZ;
    const int h = blockIdx.x % H_SZ;
    const int tid = threadIdx.x;

    __shared__ float sQn[NOPE_D];
    sQn[tid] = q_nope[((size_t)(b * H_SZ + h)) * NOPE_D + tid];
    __syncthreads();

    unsigned short* qout = qcomb + (size_t)(b * H_SZ + h) * DKV;

    // ---- latent projection: 512 outputs, 4 per thread, coalesced over c ----
    #pragma unroll
    for (int k = 0; k < 4; ++k) {
        const int c = tid + k * 128;
        const float* wcol = wkv_b + ((size_t)h * (NOPE_D + V_D)) * LORA + c;
        float acc = 0.f;
        #pragma unroll 8
        for (int d = 0; d < NOPE_D; ++d)
            acc += sQn[d] * wcol[(size_t)d * LORA];
        qout[c] = f2bf(acc);
    }

    // ---- RoPE ----
    if (tid < ROPE_D) {
        const int i = tid;
        const int p = position_ids[b];          // QLEN == 1
        const float ci = cos_t[(size_t)p * ROPE_D + i];
        const float si = sin_t[(size_t)p * ROPE_D + i];
        const float* qp = q_pe + (size_t)(b * H_SZ + h) * ROPE_D;
        float val;
        if (i < 32) val = qp[2 * i] * ci - qp[2 * i + 1] * si;
        else { int j = i - 32; val = qp[2 * j + 1] * ci + qp[2 * j] * si; }
        qout[LORA + i] = f2bf(val);

        if (h == 0) {
            const float* kp = k_pe + (size_t)b * ROPE_D;
            float kv;
            if (i < 32) kv = kp[2 * i] * ci - kp[2 * i + 1] * si;
            else { int j = i - 32; kv = kp[2 * j + 1] * ci + kp[2 * j] * si; }
            kpe_new[b * ROPE_D + i] = kv;
        }
    }
}

// =====================================================================
// Kernel 2: fused flash-decode MLA attention, bf16 WMMA, f32 accum.
// grid = B * NHG * SPLIT_T blocks, 128 threads (4 waves).
// =====================================================================
__global__ void mla_attn_kernel(const unsigned short* __restrict__ qcomb,
                                const float* __restrict__ ckv_cache,
                                const float* __restrict__ k_pe_cache,
                                const float* __restrict__ compressed_kv,
                                const float* __restrict__ kpe_new,
                                const float* __restrict__ attention_mask,
                                float* __restrict__ wsM,
                                float* __restrict__ wsL,
                                float* __restrict__ wsAcc) {
    const int gid = blockIdx.x;
    const int sp  = gid & (SPLIT_T - 1);
    const int hg  = (gid / SPLIT_T) & (NHG - 1);
    const int b   = gid / (SPLIT_T * NHG);
    const int bg  = b * NHG + hg;

    const int tid  = threadIdx.x;
    const int wid  = tid >> 5;
    const int lane = tid & 31;
    const int m    = lane & 15;     // N index of C/D, M index of A
    const int lh   = lane >> 4;     // lane half

    extern __shared__ char smem_raw[];
    unsigned int* sK32 = (unsigned int*)smem_raw;            // [64][292] dw
    unsigned int* sV32 = sK32 + TT * RQK;                    // 16384 dw (t-interleaved)
    unsigned int* sQ32 = sV32 + (TT / 2) * LORA;             // [16][292] dw
    unsigned int* sP32 = sQ32 + HG * RQK;                    // [16][36] dw
    float* sWMax  = (float*)(sP32 + HG * RP);                // [4][16]
    float* sWSum  = sWMax + 64;                              // [4][16]
    float* sM     = sWSum + 64;                              // [16]
    float* sL     = sM + 16;                                 // [16]
    float* sScale = sL + 16;                                 // [16]

    // ---- load Q (16 heads x 576 bf16) into padded LDS ----
    {
        const unsigned int* qsrc =
            (const unsigned int*)(qcomb + (size_t)(b * H_SZ + hg * HG) * DKV);
        for (int idx = tid; idx < HG * (DKV / 2); idx += 128) {
            int row = idx / (DKV / 2), cd = idx - row * (DKV / 2);
            sQ32[row * RQK + cd] = qsrc[row * (DKV / 2) + cd];
        }
        if (tid < HG) { sM[tid] = -1e30f; sL[tid] = 0.f; }
    }
    __syncthreads();

    v8f accPV[8];
    #pragma unroll
    for (int nt = 0; nt < 8; ++nt) accPV[nt] = (v8f){0,0,0,0,0,0,0,0};

    for (int kt = 0; kt < TILES; ++kt) {
        const int t0 = sp * T_PER_SP + kt * TT;

        // ---------------- stage 1: global -> bf16 -> LDS ----------------
        for (int it = 0; it < (TT * DKV) / (128 * 4); ++it) {
            const int base = tid * 4 + it * 512;
            const int r = base / DKV;
            const int c = base - r * DKV;
            const int tg = t0 + r;
            float4 v4;
            if (c < LORA) {
                const float* src = (tg < START_POS)
                    ? (ckv_cache + ((size_t)b * CACHE_LEN + tg) * LORA + c)
                    : (compressed_kv + (size_t)b * LORA + c);
                v4 = *(const float4*)src;
            } else {
                const int cc = c - LORA;
                const float* src = (tg < START_POS)
                    ? (k_pe_cache + ((size_t)b * CACHE_LEN + tg) * ROPE_D + cc)
                    : (kpe_new + b * ROPE_D + cc);
                v4 = *(const float4*)src;
            }
            const unsigned short h0 = f2bf(v4.x), h1 = f2bf(v4.y),
                                 h2 = f2bf(v4.z), h3 = f2bf(v4.w);
            const int rd = r * RQK + (c >> 1);
            sK32[rd]     = (unsigned)h0 | ((unsigned)h1 << 16);
            sK32[rd + 1] = (unsigned)h2 | ((unsigned)h3 << 16);
            if (c < LORA) {     // t-interleaved copy for PV B-fragments
                unsigned short* sVh = (unsigned short*)sV32;
                const int e = (((r >> 1) * LORA) + c) * 2 + (r & 1);
                sVh[e] = h0; sVh[e + 2] = h1; sVh[e + 4] = h2; sVh[e + 6] = h3;
            }
        }
        __syncthreads();                                  // (a) tile ready

        // ---- prefetch next tile's ckv rows into L2 (global_prefetch_b8) ----
        if (kt + 1 < TILES) {
            const int t0n = t0 + TT;
            #pragma unroll
            for (int pf = 0; pf < 72; pf += 18) {
                const int base = tid * 4 + pf * 512;
                const int r = base / DKV;
                const int c = base - r * DKV;
                const int tg = t0n + r;
                if (c < LORA && tg < START_POS)
                    __builtin_prefetch(ckv_cache + ((size_t)b * CACHE_LEN + tg) * LORA + c, 0, 1);
            }
        }

        // ---------------- stage 2: QK^T logits (wave -> 16 timesteps) ----
        v8f acc = (v8f){0,0,0,0,0,0,0,0};
        const int trow = (wid << 4) + m;
        for (int kc = 0; kc < DKV / 32; ++kc) {
            // A-frag dwords: m*RQK + kc*16 + lh*4 + {0..3}, +8+{0..3}
            // B-frag dwords: trow*RQK + kc*16 + lh*4 + {0..3}, +8+{0..3}   (all 16B aligned)
            Frag a, bq;
            const unsigned int* aB = sQ32 + m * RQK + kc * 16 + (lh << 2);
            a.q[0] = *(const u32x4*)(aB);
            a.q[1] = *(const u32x4*)(aB + 8);
            const unsigned int* bB = sK32 + trow * RQK + kc * 16 + (lh << 2);
            bq.q[0] = *(const u32x4*)(bB);
            bq.q[1] = *(const u32x4*)(bB + 8);
            acc = __builtin_amdgcn_wmma_f32_16x16x32_bf16(
                      false, a.v, false, bq.v, (short)0, acc, false, false);
        }

        const float mk = attention_mask[(size_t)b * T_TOTAL + (t0 + trow)];
        float fr[8];
        #pragma unroll
        for (int r = 0; r < 8; ++r) fr[r] = acc[r] * SCALE_F + mk;

        // per-head max over this wave's 16 timesteps (reduce inside lane-half)
        float mx[8];
        #pragma unroll
        for (int r = 0; r < 8; ++r) {
            float v = fr[r];
            v = fmaxf(v, __shfl_xor(v, 1, 32));
            v = fmaxf(v, __shfl_xor(v, 2, 32));
            v = fmaxf(v, __shfl_xor(v, 4, 32));
            v = fmaxf(v, __shfl_xor(v, 8, 32));
            mx[r] = v;
        }
        #pragma unroll
        for (int r = 0; r < 8; ++r) {
            if (lane == 0)  sWMax[wid * HG + r]     = mx[r];
            if (lane == 16) sWMax[wid * HG + 8 + r] = mx[r];
        }
        __syncthreads();                                  // (b)

        if (wid == 0 && lane < HG) {
            float tm = sWMax[lane];
            tm = fmaxf(tm, sWMax[16 + lane]);
            tm = fmaxf(tm, sWMax[32 + lane]);
            tm = fmaxf(tm, sWMax[48 + lane]);
            const float mo = sM[lane];
            const float mn = fmaxf(mo, tm);
            sScale[lane] = (mo > -1e29f) ? __expf(mo - mn) : 0.f;
            sM[lane] = mn;
        }
        __syncthreads();                                  // (c)

        // ---------------- softmax numerator + P (bf16) -------------------
        unsigned short* sPh = (unsigned short*)sP32;
        float sm[8];
        #pragma unroll
        for (int r = 0; r < 8; ++r) {
            const float p = __expf(fr[r] - sM[r + 8 * lh]);
            fr[r] = p;
            float v = p;
            v += __shfl_xor(v, 1, 32);
            v += __shfl_xor(v, 2, 32);
            v += __shfl_xor(v, 4, 32);
            v += __shfl_xor(v, 8, 32);
            sm[r] = v;
            sPh[(r + 8 * lh) * (2 * RP) + (wid << 4) + m] = f2bf(p);
        }
        #pragma unroll
        for (int r = 0; r < 8; ++r) {
            if (lane == 0)  sWSum[wid * HG + r]     = sm[r];
            if (lane == 16) sWSum[wid * HG + 8 + r] = sm[r];
        }
        __syncthreads();                                  // (d)

        if (wid == 0 && lane < HG) {
            const float ts = sWSum[lane] + sWSum[16 + lane] +
                             sWSum[32 + lane] + sWSum[48 + lane];
            sL[lane] = sL[lane] * sScale[lane] + ts;
        }

        // ---------------- stage 3: PV (wave -> 128 latent cols) ----------
        float rs[8];
        #pragma unroll
        for (int r = 0; r < 8; ++r) rs[r] = sScale[r + 8 * lh];
        #pragma unroll
        for (int nt = 0; nt < 8; ++nt)
            #pragma unroll
            for (int r = 0; r < 8; ++r)
                accPV[nt][r] *= rs[r];

        #pragma unroll
        for (int kk = 0; kk < 2; ++kk) {
            // A-frag dwords: m*RP + kk*16 + lh*4 + {0..3}, +8+{0..3} (16B aligned)
            Frag a;
            const unsigned int* pB = sP32 + m * RP + kk * 16 + (lh << 2);
            a.q[0] = *(const u32x4*)(pB);
            a.q[1] = *(const u32x4*)(pB + 8);
            #pragma unroll
            for (int nt = 0; nt < 8; ++nt) {
                Frag bq;
                const int cidx = wid * 128 + nt * 16 + m;
                #pragma unroll
                for (int v = 0; v < 8; ++v) {
                    const int tp = kk * 16 + lh * 4 + (v & 3) + ((v >> 2) << 3);   // t-pair index
                    bq.u[v] = sV32[tp * LORA + cidx];
                }
                accPV[nt] = __builtin_amdgcn_wmma_f32_16x16x32_bf16(
                                false, a.v, false, bq.v, (short)0, accPV[nt],
                                false, false);
            }
        }
        __syncthreads();                                  // (e) tile done
    }

    // ---------------- write split partials ----------------
    if (tid < HG) {
        wsM[(bg * SPLIT_T + sp) * HG + tid] = sM[tid];
        wsL[(bg * SPLIT_T + sp) * HG + tid] = sL[tid];
    }
    float* aout = wsAcc + ((size_t)(bg * SPLIT_T + sp) * HG) * LORA;
    #pragma unroll
    for (int nt = 0; nt < 8; ++nt)
        #pragma unroll
        for (int r = 0; r < 8; ++r)
            aout[(size_t)(r + 8 * lh) * LORA + wid * 128 + nt * 16 + m] = accPV[nt][r];
}

// =====================================================================
// Kernel 3: combine T-splits, normalize, project with w_uv (128x512).
// grid = B*H blocks, 256 threads (8 waves).
// =====================================================================
__global__ void combine_project_kernel(const float* __restrict__ wsM,
                                       const float* __restrict__ wsL,
                                       const float* __restrict__ wsAcc,
                                       const float* __restrict__ wkv_b,
                                       float* __restrict__ out) {
    const int b  = blockIdx.x / H_SZ;
    const int h  = blockIdx.x % H_SZ;
    const int hg = h >> 4;
    const int hl = h & 15;
    const int bg = b * NHG + hg;
    const int tid = threadIdx.x;

    __shared__ float sX[LORA];
    __shared__ float sW[SPLIT_T];
    __shared__ float sInvDen;

    if (tid == 0) {
        const float m0 = wsM[(bg * SPLIT_T + 0) * HG + hl];
        const float m1 = wsM[(bg * SPLIT_T + 1) * HG + hl];
        const float ms = fmaxf(m0, m1);
        const float w0 = __expf(m0 - ms);
        const float w1 = __expf(m1 - ms);
        const float den = wsL[(bg * SPLIT_T + 0) * HG + hl] * w0 +
                          wsL[(bg * SPLIT_T + 1) * HG + hl] * w1;
        sW[0] = w0; sW[1] = w1;
        sInvDen = 1.0f / den;
    }
    __syncthreads();

    const float* a0 = wsAcc + ((size_t)(bg * SPLIT_T + 0) * HG + hl) * LORA;
    const float* a1 = wsAcc + ((size_t)(bg * SPLIT_T + 1) * HG + hl) * LORA;
    for (int c = tid; c < LORA; c += 256)
        sX[c] = (sW[0] * a0[c] + sW[1] * a1[c]) * sInvDen;
    __syncthreads();

    // out[d] = sum_c x[c] * w_uv[h, d, c]; 8 waves x 16 d each, coalesced rows
    const int wid = tid >> 5, lane = tid & 31;
    #pragma unroll 4
    for (int dd = 0; dd < 16; ++dd) {
        const int d = wid * 16 + dd;
        const float* wrow = wkv_b + ((size_t)h * (NOPE_D + V_D) + NOPE_D + d) * LORA;
        float a = 0.f;
        #pragma unroll
        for (int k = 0; k < LORA / 32; ++k)
            a += wrow[lane + k * 32] * sX[lane + k * 32];
        a += __shfl_xor(a, 16, 32);
        a += __shfl_xor(a, 8, 32);
        a += __shfl_xor(a, 4, 32);
        a += __shfl_xor(a, 2, 32);
        a += __shfl_xor(a, 1, 32);
        if (lane == 0) out[((size_t)b * H_SZ + h) * V_D + d] = a;
    }
}

// =====================================================================
extern "C" void kernel_launch(void* const* d_in, const int* in_sizes, int n_in,
                              void* d_out, int out_size, void* d_ws, size_t ws_size,
                              hipStream_t stream) {
    const float* q_nope         = (const float*)d_in[0];
    const float* q_pe           = (const float*)d_in[1];
    const float* compressed_kv  = (const float*)d_in[2];
    const float* k_pe           = (const float*)d_in[3];
    const int*   position_ids   = (const int*)d_in[4];
    const float* attention_mask = (const float*)d_in[5];
    /* d_in[6] = start_pos (compile-time constant START_POS) */
    const float* ckv_cache      = (const float*)d_in[7];
    const float* k_pe_cache     = (const float*)d_in[8];
    const float* sin_t          = (const float*)d_in[9];
    const float* cos_t          = (const float*)d_in[10];
    const float* wkv_b          = (const float*)d_in[11];
    float* out = (float*)d_out;

    char* ws = (char*)d_ws;
    unsigned short* qcomb = (unsigned short*)(ws + OFF_QCOMB);
    float* kpe_new = (float*)(ws + OFF_KPEN);
    float* wsM     = (float*)(ws + OFF_M);
    float* wsL     = (float*)(ws + OFF_L);
    float* wsAcc   = (float*)(ws + OFF_ACC);

    prep_q_kernel<<<dim3(B_SZ * H_SZ), dim3(128), 0, stream>>>(
        q_nope, q_pe, k_pe, position_ids, sin_t, cos_t, wkv_b, qcomb, kpe_new);

    // dynamic LDS: sK(74752) + sV(65536) + sQ(18688) + sP(2304) + scratch(704)
    const size_t smem = (size_t)(TT * RQK + (TT / 2) * LORA + HG * RQK + HG * RP) * 4
                        + (64 + 64 + 16 + 16 + 16) * 4;
    mla_attn_kernel<<<dim3(B_SZ * NHG * SPLIT_T), dim3(128), smem, stream>>>(
        qcomb, ckv_cache, k_pe_cache, compressed_kv, kpe_new, attention_mask,
        wsM, wsL, wsAcc);

    combine_project_kernel<<<dim3(B_SZ * H_SZ), dim3(256), 0, stream>>>(
        wsM, wsL, wsAcc, wkv_b, out);
}
